// TinyTransformer_6330781794638
// MI455X (gfx1250) — matrix operand order repaired and checked by reading, not compiled, not run on hardware
//
#include <hip/hip_runtime.h>

// ---------------------------------------------------------------------------
// TinyTransformer forward for MI455X (gfx1250, wave32, WMMA).
// All GEMMs run as bf16 WMMA (v_wmma_f32_16x16x32_bf16) with f32 accumulate.
// Softmax / LayerNorm / residuals stay f32. Weights are pre-transposed once
// into [K][N] bf16; each lane gathers its WMMA B column with strided 2B loads.
// GEMM K-loop is unrolled x2 with ping-pong fragment buffers (no register
// rotation copies); global_load_b128 latency overlaps WMMA issue per wave.
// Workspace requirement: ~300 MB (laid out below).
// ---------------------------------------------------------------------------

typedef __bf16 bf16_t;
typedef __attribute__((ext_vector_type(16))) __bf16 v16bf;
typedef __attribute__((ext_vector_type(8)))  __bf16 v8bf;
typedef __attribute__((ext_vector_type(8)))  float  v8f;

#define D_MODEL 512
#define SEQ     2048
#define BATCH   4
#define HID     2048
#define VOCABSZ 32000
#define TOKENS  (BATCH * SEQ)   // 8192

// ---------------- embedding: word + positional, f32 + bf16 copies ----------
__global__ __launch_bounds__(256)
void emb_kernel(const int* __restrict__ x, const float* __restrict__ wemb,
                const float* __restrict__ pemb,
                float* __restrict__ emb_f, bf16_t* __restrict__ emb_b) {
  const int  row = blockIdx.x;           // b*SEQ + s
  const int  s   = row & (SEQ - 1);
  const long tok = (long)x[row];
  const float* wp = wemb + tok * D_MODEL;
  const float* pp = pemb + (long)s * D_MODEL;
  float*  of = emb_f + (long)row * D_MODEL;
  bf16_t* ob = emb_b + (long)row * D_MODEL;
  for (int d = threadIdx.x; d < D_MODEL; d += 256) {
    float v = wp[d] + pp[d];
    of[d] = v;
    ob[d] = (bf16_t)v;
  }
}

// ---------------- transpose + convert to bf16 ([rows][cols] -> [cols][rows])
template <typename T>
__global__ __launch_bounds__(256)
void transpose_to_bf16_kernel(const T* __restrict__ in, bf16_t* __restrict__ out,
                              int rows, int cols) {
  __shared__ float tile[32][33];
  const int r0 = blockIdx.y * 32, c0 = blockIdx.x * 32;
  const int tx = threadIdx.x, ty = threadIdx.y;   // block = 32 x 8
  #pragma unroll
  for (int i = ty; i < 32; i += 8)
    tile[i][tx] = (float)in[(long)(r0 + i) * cols + (c0 + tx)];
  __syncthreads();
  #pragma unroll
  for (int i = ty; i < 32; i += 8)
    out[(long)(c0 + i) * rows + (r0 + tx)] = (bf16_t)tile[tx][i];
}

// ---------------- fragment loader ------------------------------------------
// A fragments (16x32 bf16, ISA layout): lane<16 -> row m0+lane, K chunks
// {0..7},{16..23}; lane>=16 -> row m0+lane-16, K chunks {8..15},{24..31}.
// B fragment: lane holds column n0+(lane&15), K in the same chunks as A (strided gather).
__device__ __forceinline__ void load_frags(const bf16_t* a0p, const bf16_t* a1p,
                                           const bf16_t* bp, long ldb,
                                           v16bf& a0, v16bf& a1,
                                           v16bf& b0, v16bf& b1) {
  v8bf a0lo = *(const v8bf*)(a0p);
  v8bf a0hi = *(const v8bf*)(a0p + 16);
  v8bf a1lo = *(const v8bf*)(a1p);
  v8bf a1hi = *(const v8bf*)(a1p + 16);
  a0 = __builtin_shufflevector(a0lo, a0hi, 0,1,2,3,4,5,6,7,8,9,10,11,12,13,14,15);
  a1 = __builtin_shufflevector(a1lo, a1hi, 0,1,2,3,4,5,6,7,8,9,10,11,12,13,14,15);
  #pragma unroll
  for (int s = 0; s < 16; ++s) {
    const long ko = (long)(s < 8 ? s : s + 8) * ldb;
    b0[s] = bp[ko];
    b1[s] = bp[ko + 16];
  }
}

// ---------------- generic bf16 WMMA GEMM -----------------------------------
// C[m,n] = act( A[m,:] . B[:,n] + bias[n] ),  A:[M,K] bf16, B:[K][N] bf16.
// Block = 256 thr = 8 waves -> 128(M) x 64(N) tile; wave tile 32x32 (4 accs).
// K must be a multiple of 64 (all GEMMs here have K in {512, 2048}).
// causal=1: skip blocks entirely above the diagonal (q-tile < key-tile).
__global__ __launch_bounds__(256)
void gemm_bf16_wmma_kernel(const bf16_t* __restrict__ A, const bf16_t* __restrict__ B,
                           const float* __restrict__ bias,
                           float* __restrict__ Cf, bf16_t* __restrict__ Cb,
                           int N, int K,
                           long strideA, long strideB, long strideC,
                           int relu, int causal) {
  if (causal && (int)(blockIdx.x * 64) > (int)(blockIdx.y * 128 + 127)) return;

  const int lane = threadIdx.x & 31;
  const int wave = threadIdx.x >> 5;
  const int m0 = blockIdx.y * 128 + (wave >> 1) * 32;
  const int n0 = blockIdx.x * 64  + (wave & 1) * 32;
  const long zb = blockIdx.z;

  const bf16_t* Ab = A + zb * strideA;
  const bf16_t* Bb = B + zb * strideB;

  const int khalf = lane >> 4;
  const int arow  = lane & 15;
  const bf16_t* a0p = Ab + (long)(m0 + arow) * K + khalf * 8;
  const bf16_t* a1p = a0p + (long)16 * K;
  const bf16_t* b0p = Bb + (long)(khalf * 8) * N + n0 + arow;

  v8f acc00 = {}, acc01 = {}, acc10 = {}, acc11 = {};

  // ping-pong double buffering: X holds k0, Y holds k0+32; fixed registers,
  // no rotation copies. Loads for one buffer fly while the other computes.
  v16bf a0x, a1x, b0x, b1x, a0y, a1y, b0y, b1y;
  load_frags(a0p, a1p, b0p, N, a0x, a1x, b0x, b1x);

  for (int k0 = 0; k0 < K; k0 += 64) {
    // stage Y <- k0+32 (always in range: K % 64 == 0)
    load_frags(a0p + k0 + 32, a1p + k0 + 32, b0p + (long)(k0 + 32) * N, N,
               a0y, a1y, b0y, b1y);
    __builtin_prefetch(a0p + k0 + 64, 0, 1);   // speculative, dropped if OOB
    __builtin_prefetch(a1p + k0 + 64, 0, 1);
    __builtin_prefetch(b0p + (long)(k0 + 64) * N, 0, 1);

    acc00 = __builtin_amdgcn_wmma_f32_16x16x32_bf16(false, a0x, false, b0x, (short)0, acc00, false, false);
    acc01 = __builtin_amdgcn_wmma_f32_16x16x32_bf16(false, a0x, false, b1x, (short)0, acc01, false, false);
    acc10 = __builtin_amdgcn_wmma_f32_16x16x32_bf16(false, a1x, false, b0x, (short)0, acc10, false, false);
    acc11 = __builtin_amdgcn_wmma_f32_16x16x32_bf16(false, a1x, false, b1x, (short)0, acc11, false, false);

    if (k0 + 64 < K)  // stage X <- k0+64 while Y computes
      load_frags(a0p + k0 + 64, a1p + k0 + 64, b0p + (long)(k0 + 64) * N, N,
                 a0x, a1x, b0x, b1x);

    acc00 = __builtin_amdgcn_wmma_f32_16x16x32_bf16(false, a0y, false, b0y, (short)0, acc00, false, false);
    acc01 = __builtin_amdgcn_wmma_f32_16x16x32_bf16(false, a0y, false, b1y, (short)0, acc01, false, false);
    acc10 = __builtin_amdgcn_wmma_f32_16x16x32_bf16(false, a1y, false, b0y, (short)0, acc10, false, false);
    acc11 = __builtin_amdgcn_wmma_f32_16x16x32_bf16(false, a1y, false, b1y, (short)0, acc11, false, false);
  }

  // C/D layout: VGPR i -> row rb+i, col = lane&15 (rb = 8 for lanes 16..31)
  const int col = lane & 15;
  const int rb  = khalf * 8;
  const int nA = n0 + col, nB = n0 + 16 + col;
  const float biasA = bias ? bias[nA] : 0.f;
  const float biasB = bias ? bias[nB] : 0.f;
  float*  cf = Cf ? Cf + zb * strideC : (float*)0;
  bf16_t* cb = Cb ? Cb + zb * strideC : (bf16_t*)0;
  #pragma unroll
  for (int i = 0; i < 8; ++i) {
    const long mA = (long)m0 + rb + i;
    const long mB = mA + 16;
    float v00 = acc00[i] + biasA;
    float v01 = acc01[i] + biasB;
    float v10 = acc10[i] + biasA;
    float v11 = acc11[i] + biasB;
    if (relu) {
      v00 = fmaxf(v00, 0.f); v01 = fmaxf(v01, 0.f);
      v10 = fmaxf(v10, 0.f); v11 = fmaxf(v11, 0.f);
    }
    const long i00 = mA * N + nA, i01 = mA * N + nB;
    const long i10 = mB * N + nA, i11 = mB * N + nB;
    if (cf) { cf[i00] = v00; cf[i01] = v01; cf[i10] = v10; cf[i11] = v11; }
    if (cb) { cb[i00] = (bf16_t)v00; cb[i01] = (bf16_t)v01;
              cb[i10] = (bf16_t)v10; cb[i11] = (bf16_t)v11; }
  }
}

// ---------------- block reduction (256 threads = 8 waves, wave32) ----------
__device__ __forceinline__ float block_reduce(float v, int is_max, float* red) {
  #pragma unroll
  for (int off = 16; off > 0; off >>= 1) {
    float o = __shfl_xor(v, off, 32);
    v = is_max ? fmaxf(v, o) : (v + o);
  }
  const int tid = threadIdx.x;
  if ((tid & 31) == 0) red[tid >> 5] = v;
  __syncthreads();
  if (tid == 0) {
    float r = red[0];
    for (int i = 1; i < 8; ++i) r = is_max ? fmaxf(r, red[i]) : (r + red[i]);
    red[0] = r;
  }
  __syncthreads();
  float r = red[0];
  __syncthreads();
  return r;
}

// ---------------- causal softmax: f32 scores row -> bf16 attn row ----------
__global__ __launch_bounds__(256)
void softmax_causal_kernel(const float* __restrict__ scores, bf16_t* __restrict__ attn) {
  __shared__ float red[8];
  const long row    = blockIdx.x;                 // b*SEQ + q
  const int  q      = (int)(row & (SEQ - 1));
  const int  nvalid = q + 1;
  const float scale = 0.04419417382415922f;       // 1/sqrt(512)
  const float* sp = scores + row * (long)SEQ;
  bf16_t*      ap = attn   + row * (long)SEQ;
  const int tid = threadIdx.x;

  float mx = -3.4e38f;
  for (int j = tid; j < nvalid; j += 256) mx = fmaxf(mx, sp[j] * scale);
  mx = block_reduce(mx, 1, red);

  float s = 0.f;
  for (int j = tid; j < nvalid; j += 256) s += __expf(sp[j] * scale - mx);
  s = block_reduce(s, 0, red);
  const float inv = 1.f / s;

  for (int j = tid; j < SEQ; j += 256)
    ap[j] = (j < nvalid) ? (bf16_t)(__expf(sp[j] * scale - mx) * inv) : (bf16_t)0.f;
}

// ---------------- residual add + LayerNorm (d=512, 256 thr x 2 elems) ------
__global__ __launch_bounds__(256)
void add_ln_kernel(const float* __restrict__ xa, const float* __restrict__ xb,
                   const float* __restrict__ g, const float* __restrict__ b,
                   float* __restrict__ outf, bf16_t* __restrict__ outb) {
  __shared__ float red[8];
  const long row = blockIdx.x;
  const float* pa = xa + row * D_MODEL;
  const float* pb = xb + row * D_MODEL;
  const int t = threadIdx.x;
  const float v0 = pa[t] + pb[t];
  const float v1 = pa[t + 256] + pb[t + 256];
  const float mu  = block_reduce(v0 + v1, 0, red) * (1.f / D_MODEL);
  const float d0 = v0 - mu, d1 = v1 - mu;
  const float var = block_reduce(d0 * d0 + d1 * d1, 0, red) * (1.f / D_MODEL);
  const float rs = rsqrtf(var + 1e-5f);
  const float o0 = d0 * rs * g[t] + b[t];
  const float o1 = d1 * rs * g[t + 256] + b[t + 256];
  if (outf) { float* p = outf + row * D_MODEL; p[t] = o0; p[t + 256] = o1; }
  if (outb) { bf16_t* p = outb + row * D_MODEL; p[t] = (bf16_t)o0; p[t + 256] = (bf16_t)o1; }
}

// ---------------------------------------------------------------------------
extern "C" void kernel_launch(void* const* d_in, const int* in_sizes, int n_in,
                              void* d_out, int out_size, void* d_ws, size_t ws_size,
                              hipStream_t stream) {
  (void)in_sizes; (void)n_in; (void)out_size; (void)ws_size;
  const int*   x       = (const int*)  d_in[0];
  const float* wemb    = (const float*)d_in[1];
  const float* pemb    = (const float*)d_in[2];
  const float* Wq      = (const float*)d_in[3];
  const float* Wk      = (const float*)d_in[4];
  const float* Wv      = (const float*)d_in[5];
  const float* g1      = (const float*)d_in[6];
  const float* b1      = (const float*)d_in[7];
  const float* w_ffn1  = (const float*)d_in[8];
  const float* b_ffn1  = (const float*)d_in[9];
  const float* w_ffn2  = (const float*)d_in[10];
  const float* b_ffn2  = (const float*)d_in[11];
  const float* g2      = (const float*)d_in[12];
  const float* b2      = (const float*)d_in[13];
  const float* w_vocab = (const float*)d_in[14];
  const float* b_vocab = (const float*)d_in[15];
  float* out = (float*)d_out;

  char* ws = (char*)d_ws;
  size_t cur = 0;
  auto alloc = [&](size_t bytes) -> void* {
    void* p = ws + cur;
    cur += (bytes + 255) & ~(size_t)255;
    return p;
  };
  bf16_t* Wqt    = (bf16_t*)alloc((size_t)D_MODEL * D_MODEL * 2);
  bf16_t* Wkt    = (bf16_t*)alloc((size_t)D_MODEL * D_MODEL * 2);
  bf16_t* Wvt    = (bf16_t*)alloc((size_t)D_MODEL * D_MODEL * 2);
  bf16_t* Wf1t   = (bf16_t*)alloc((size_t)D_MODEL * HID * 2);      // [512][2048]
  bf16_t* Wf2t   = (bf16_t*)alloc((size_t)HID * D_MODEL * 2);      // [2048][512]
  bf16_t* Wvoct  = (bf16_t*)alloc((size_t)D_MODEL * VOCABSZ * 2);  // [512][32000]
  float*  emb_f  = (float*) alloc((size_t)TOKENS * D_MODEL * 4);
  bf16_t* emb_b  = (bf16_t*)alloc((size_t)TOKENS * D_MODEL * 2);
  bf16_t* Qb     = (bf16_t*)alloc((size_t)TOKENS * D_MODEL * 2);
  bf16_t* Kb     = (bf16_t*)alloc((size_t)TOKENS * D_MODEL * 2);
  bf16_t* Ktb    = (bf16_t*)alloc((size_t)TOKENS * D_MODEL * 2);   // per-batch [512][2048]
  bf16_t* Vb     = (bf16_t*)alloc((size_t)TOKENS * D_MODEL * 2);
  float*  scores = (float*) alloc((size_t)BATCH * SEQ * SEQ * 4);
  bf16_t* attn   = (bf16_t*)alloc((size_t)BATCH * SEQ * SEQ * 2);
  float*  ctx    = (float*) alloc((size_t)TOKENS * D_MODEL * 4);
  float*  out1f  = (float*) alloc((size_t)TOKENS * D_MODEL * 4);
  bf16_t* out1b  = (bf16_t*)alloc((size_t)TOKENS * D_MODEL * 2);
  bf16_t* hid    = (bf16_t*)alloc((size_t)TOKENS * HID * 2);
  float*  ffn2f  = (float*) alloc((size_t)TOKENS * D_MODEL * 4);
  bf16_t* finalb = (bf16_t*)alloc((size_t)TOKENS * D_MODEL * 2);

  const dim3 tb(32, 8);

  // 1) one-time weight transposes (f32 [N][K] -> bf16 [K][N])
  transpose_to_bf16_kernel<float><<<dim3(D_MODEL/32, D_MODEL/32), tb, 0, stream>>>(Wq, Wqt, D_MODEL, D_MODEL);
  transpose_to_bf16_kernel<float><<<dim3(D_MODEL/32, D_MODEL/32), tb, 0, stream>>>(Wk, Wkt, D_MODEL, D_MODEL);
  transpose_to_bf16_kernel<float><<<dim3(D_MODEL/32, D_MODEL/32), tb, 0, stream>>>(Wv, Wvt, D_MODEL, D_MODEL);
  transpose_to_bf16_kernel<float><<<dim3(D_MODEL/32, HID/32),     tb, 0, stream>>>(w_ffn1, Wf1t, HID, D_MODEL);
  transpose_to_bf16_kernel<float><<<dim3(HID/32, D_MODEL/32),     tb, 0, stream>>>(w_ffn2, Wf2t, D_MODEL, HID);
  transpose_to_bf16_kernel<float><<<dim3(D_MODEL/32, VOCABSZ/32), tb, 0, stream>>>(w_vocab, Wvoct, VOCABSZ, D_MODEL);

  // 2) embeddings
  emb_kernel<<<TOKENS, 256, 0, stream>>>(x, wemb, pemb, emb_f, emb_b);

  // 3) Q / K / V projections (bf16 out)
  gemm_bf16_wmma_kernel<<<dim3(D_MODEL/64, TOKENS/128, 1), 256, 0, stream>>>(
      emb_b, Wqt, (const float*)0, (float*)0, Qb, D_MODEL, D_MODEL, 0, 0, 0, 0, 0);
  gemm_bf16_wmma_kernel<<<dim3(D_MODEL/64, TOKENS/128, 1), 256, 0, stream>>>(
      emb_b, Wkt, (const float*)0, (float*)0, Kb, D_MODEL, D_MODEL, 0, 0, 0, 0, 0);
  gemm_bf16_wmma_kernel<<<dim3(D_MODEL/64, TOKENS/128, 1), 256, 0, stream>>>(
      emb_b, Wvt, (const float*)0, (float*)0, Vb, D_MODEL, D_MODEL, 0, 0, 0, 0, 0);

  // 4) K -> K^T per batch ([2048][512] -> [512][2048]) for the scores B operand
  for (int bz = 0; bz < BATCH; ++bz)
    transpose_to_bf16_kernel<bf16_t><<<dim3(D_MODEL/32, SEQ/32), tb, 0, stream>>>(
        Kb + (long)bz * SEQ * D_MODEL, Ktb + (long)bz * D_MODEL * SEQ, SEQ, D_MODEL);

  // 5) scores = Q @ K^T (f32 out, causal block skip)
  gemm_bf16_wmma_kernel<<<dim3(SEQ/64, SEQ/128, BATCH), 256, 0, stream>>>(
      Qb, Ktb, (const float*)0, scores, (bf16_t*)0, SEQ, D_MODEL,
      (long)SEQ * D_MODEL, (long)D_MODEL * SEQ, (long)SEQ * SEQ, 0, 1);

  // 6) causal softmax (scale fused), writes zero-padded bf16 attn
  softmax_causal_kernel<<<TOKENS, 256, 0, stream>>>(scores, attn);

  // 7) context = attn @ V (V already [key][d] = [K][N])
  gemm_bf16_wmma_kernel<<<dim3(D_MODEL/64, SEQ/128, BATCH), 256, 0, stream>>>(
      attn, Vb, (const float*)0, ctx, (bf16_t*)0, D_MODEL, SEQ,
      (long)SEQ * SEQ, (long)SEQ * D_MODEL, (long)SEQ * D_MODEL, 0, 0);

  // 8) out1 = LN(emb + ctx)
  add_ln_kernel<<<TOKENS, 256, 0, stream>>>(emb_f, ctx, g1, b1, out1f, out1b);

  // 9) FFN1 with fused bias + ReLU (bf16 out)
  gemm_bf16_wmma_kernel<<<dim3(HID/64, TOKENS/128, 1), 256, 0, stream>>>(
      out1b, Wf1t, b_ffn1, (float*)0, hid, HID, D_MODEL, 0, 0, 0, 1, 0);

  // 10) FFN2 with fused bias (f32 out)
  gemm_bf16_wmma_kernel<<<dim3(D_MODEL/64, TOKENS/128, 1), 256, 0, stream>>>(
      hid, Wf2t, b_ffn2, ffn2f, (bf16_t*)0, D_MODEL, HID, 0, 0, 0, 0, 0);

  // 11) final = LN(out1 + ffn) (bf16 only, feeds vocab GEMM)
  add_ln_kernel<<<TOKENS, 256, 0, stream>>>(out1f, ffn2f, g2, b2, (float*)0, finalb);

  // 12) logits = final @ w_vocab^T + b_vocab, streamed straight to d_out
  gemm_bf16_wmma_kernel<<<dim3(VOCABSZ/64, TOKENS/128, 1), 256, 0, stream>>>(
      finalb, Wvoct, b_vocab, out, (bf16_t*)0, VOCABSZ, D_MODEL, 0, 0, 0, 0, 0);
}
